// AttentionBlock_57243324121226
// MI455X (gfx1250) — compile-verified, matrix operand order checked by reading, not run on hardware
//
#include <hip/hip_runtime.h>
#include <hip/hip_bf16.h>

// ---------------------------------------------------------------------------
// Transformer block (pre-LN, causal MHA + MLP) for MI455X (gfx1250, wave32).
// Matmuls on v_wmma_f32_16x16x32_bf16 (f32 accum); global->LDS staging uses
// CDNA5 async copies (global_load_async_to_lds_b128 + s_wait_asynccnt) with
// double-buffered LDS so copies overlap WMMA compute.
// ---------------------------------------------------------------------------

#define BB 4
#define TT 2048
#define CC 1024
#define HH 16
#define DD 64
#define MM (BB * TT)   // 8192 rows

typedef __attribute__((ext_vector_type(16))) __bf16 v16bf;
typedef __attribute__((ext_vector_type(8)))  float  v8f;
typedef __attribute__((address_space(3))) unsigned short lds_us;

union FragBF { v16bf v; unsigned int u[8]; unsigned short h[16]; };
union FragF  { v8f   v; float f[8]; };

__device__ __forceinline__ unsigned short f32_to_bf16(float f) {
  union { float f; unsigned u; } cv; cv.f = f;
  unsigned x = cv.u;
  unsigned r = x + 0x7fffu + ((x >> 16) & 1u);   // round-to-nearest-even
  return (unsigned short)(r >> 16);
}

// Async copy 16 bytes global -> LDS, tracked by ASYNCcnt (CDNA5 TDM-lite path).
__device__ __forceinline__ void async_b128(const unsigned short* gaddr, unsigned short* lptr) {
  asm volatile("global_load_async_to_lds_b128 %0, %1, off"
               :: "v"((lds_us*)lptr), "v"(gaddr) : "memory");
}
__device__ __forceinline__ void wait_async0() {
  asm volatile("s_wait_asynccnt 0" ::: "memory");
}

// Load a 16-lane-striped 16x32 bf16 fragment (A- or B-operand) whose K-pairs
// are contiguous starting at `base` (per-lane row/col pointer).
// ISA 16-bit layout: vgpr v holds K = koff + (v&3)*2 + (v>>2)*16 (+1 packed),
// koff = (lane>=16) ? 8 : 0.
__device__ __forceinline__ void load_frag16(FragBF& fr, const unsigned short* base, int koff) {
#pragma unroll
  for (int v = 0; v < 8; ++v) {
    int k = koff + ((v & 3) << 1) + ((v >> 2) << 4);
    fr.u[v] = *(const unsigned int*)(base + k);
  }
}

// ---------------------------------------------------------------------------
// Weight packing / conversion kernels
// ---------------------------------------------------------------------------

// wq/wk/wv [H][C][D] f32 -> fused bf16 [C][3*H*D]  (q | k | v blocks of 1024)
__global__ __launch_bounds__(256)
void pack_qkv_kernel(const float* __restrict__ wq, const float* __restrict__ wk,
                     const float* __restrict__ wv, unsigned short* __restrict__ out) {
  int i = blockIdx.x * 256 + threadIdx.x;
  const int per = HH * CC * DD;            // 1,048,576
  if (i >= 3 * per) return;
  int sel = i / per;
  int r = i - sel * per;
  int h = r / (CC * DD);
  int c = (r / DD) % CC;
  int d = r % DD;
  const float* src = (sel == 0) ? wq : (sel == 1) ? wk : wv;
  float v = src[(size_t)h * CC * DD + (size_t)c * DD + d];
  out[(size_t)c * (3 * CC) + sel * CC + h * DD + d] = f32_to_bf16(v);
}

__global__ __launch_bounds__(256)
void convert_kernel(const float* __restrict__ in, unsigned short* __restrict__ out, int n) {
  int i = blockIdx.x * 256 + threadIdx.x;
  if (i < n) out[i] = f32_to_bf16(in[i]);
}

// ---------------------------------------------------------------------------
// LayerNorm: one row (C=1024) per block, 256 threads, bf16 output
// ---------------------------------------------------------------------------
__global__ __launch_bounds__(256)
void layernorm_kernel(const float* __restrict__ x, const float* __restrict__ sc,
                      const float* __restrict__ bi, unsigned short* __restrict__ out) {
  const int row = blockIdx.x;
  const float* xr = x + (size_t)row * CC;
  const int tid = threadIdx.x, lane = tid & 31, wave = tid >> 5;
  __shared__ float red[8];
  float v[4];
  float s = 0.0f;
#pragma unroll
  for (int i = 0; i < 4; ++i) { v[i] = xr[tid + i * 256]; s += v[i]; }
#pragma unroll
  for (int off = 16; off; off >>= 1) s += __shfl_xor(s, off, 32);
  if (lane == 0) red[wave] = s;
  __syncthreads();
  float tot = 0.0f;
#pragma unroll
  for (int i = 0; i < 8; ++i) tot += red[i];
  const float mu = tot * (1.0f / CC);
  __syncthreads();
  float q = 0.0f;
#pragma unroll
  for (int i = 0; i < 4; ++i) { float d = v[i] - mu; q += d * d; }
#pragma unroll
  for (int off = 16; off; off >>= 1) q += __shfl_xor(q, off, 32);
  if (lane == 0) red[wave] = q;
  __syncthreads();
  tot = 0.0f;
#pragma unroll
  for (int i = 0; i < 8; ++i) tot += red[i];
  const float r = rsqrtf(tot * (1.0f / CC) + 1e-6f);
#pragma unroll
  for (int i = 0; i < 4; ++i) {
    int c = tid + i * 256;
    out[(size_t)row * CC + c] = f32_to_bf16((v[i] - mu) * r * sc[c] + bi[c]);
  }
}

// ---------------------------------------------------------------------------
// WMMA GEMM: C[M,N] = A[M,K] * B[K,N] (+epilogue). Block tile 64x128, 8 waves,
// each wave owns a 16x64 strip (4 accumulators). K-loop in 32-chunks.
// Double-buffered LDS: A-panel staged via async b128 copies (ASYNCcnt),
// B-panel staged transposed synchronously; next tile's copy overlaps WMMAs.
// EPI: 0 = bf16 raw, 1 = f32 res + acc + bias, 2 = bf16 relu(acc + bias)
// ---------------------------------------------------------------------------
template <int EPI>
__global__ __launch_bounds__(256)
void gemm_bf16_kernel(const unsigned short* __restrict__ A,
                      const unsigned short* __restrict__ Bm,
                      const float* __restrict__ bias,
                      const float* __restrict__ res,
                      void* __restrict__ outv,
                      int M, int N, int K) {
  __shared__ unsigned short As[2][64][40];    // [m][k], 80B rows (16B aligned)
  __shared__ unsigned short BsT[2][128][36];  // [n][k], transposed, +4 pad
  const int m0 = blockIdx.y * 64;
  const int n0 = blockIdx.x * 128;
  const int tid = threadIdx.x;
  const int lane = tid & 31, wave = tid >> 5;
  const int waveM = wave >> 1, waveN = wave & 1;
  const int hi = lane >> 4, nlo = lane & 15;
  const int koff = hi * 8;

  FragF acc[4];
#pragma unroll
  for (int g = 0; g < 4; ++g)
#pragma unroll
    for (int i = 0; i < 8; ++i) acc[g].f[i] = 0.0f;

  // A tile 64x32 = 256 b128 transfers: one per thread.
  auto stageA = [&](int buf, int k0) {
    int row = tid >> 2, seg = tid & 3;
    async_b128(A + (size_t)(m0 + row) * K + k0 + seg * 8, &As[buf][row][seg * 8]);
  };
  // B tile 32x128 staged transposed (2048 u32 pairs, 8 per thread).
  auto stageB = [&](int buf, int k0) {
#pragma unroll
    for (int i = 0; i < 8; ++i) {
      int p = tid + i * 256;
      int row = p >> 6;              // k within chunk
      int col = (p & 63) * 2;        // n within tile
      unsigned int val = *(const unsigned int*)(Bm + (size_t)(k0 + row) * N + n0 + col);
      BsT[buf][col][row]     = (unsigned short)(val & 0xffffu);
      BsT[buf][col + 1][row] = (unsigned short)(val >> 16);
    }
  };

  stageA(0, 0);
  stageB(0, 0);
  wait_async0();
  __syncthreads();

  int buf = 0;
  for (int k0 = 0; k0 < K; k0 += 32) {
    const int nk = k0 + 32;
    if (nk < K) {            // overlap next tile's copy with this tile's WMMAs
      stageA(buf ^ 1, nk);
      stageB(buf ^ 1, nk);
    }
    if (k0 + 64 < K)         // warm L2 two chunks ahead
      __builtin_prefetch(A + (size_t)(m0 + (tid >> 2)) * K + k0 + 64 + (tid & 3) * 8, 0, 0);

    FragBF a;
    load_frag16(a, &As[buf][waveM * 16 + nlo][0], koff);
#pragma unroll
    for (int g = 0; g < 4; ++g) {
      FragBF b;
      load_frag16(b, &BsT[buf][waveN * 64 + g * 16 + nlo][0], koff);
      acc[g].v = __builtin_amdgcn_wmma_f32_16x16x32_bf16(
          false, a.v, false, b.v, (short)0, acc[g].v, false, false);
    }
    wait_async0();           // my async deposits for buf^1 done
    __syncthreads();         // all waves' ds stores + asyncs visible
    buf ^= 1;
  }

  // Epilogue. C layout: vgpr v, lanes<16 -> row v, lanes>=16 -> row v+8.
#pragma unroll
  for (int g = 0; g < 4; ++g) {
#pragma unroll
    for (int v = 0; v < 8; ++v) {
      int row = m0 + waveM * 16 + v + hi * 8;
      int col = n0 + waveN * 64 + g * 16 + nlo;
      size_t idx = (size_t)row * N + col;
      float val = acc[g].f[v];
      if (EPI == 0) {
        ((unsigned short*)outv)[idx] = f32_to_bf16(val);
      } else if (EPI == 1) {
        ((float*)outv)[idx] = res[idx] + val + bias[col];
      } else {
        float t = val + bias[col];
        ((unsigned short*)outv)[idx] = f32_to_bf16(t > 0.0f ? t : 0.0f);
      }
    }
  }
}

// ---------------------------------------------------------------------------
// Causal flash attention. Grid: (T/128, H, B), 256 threads.
// 8 waves x 16 query rows; 32-key tiles: K staged via async b128 (ASYNCcnt),
// V staged transposed; online softmax; P round-trips per-wave LDS.
// qkv layout: [B*T][3072] bf16 = [q(h*64+d) | k | v]. out: [B*T][1024] bf16.
// ---------------------------------------------------------------------------
__global__ __launch_bounds__(256)
void attention_kernel(const unsigned short* __restrict__ qkv,
                      unsigned short* __restrict__ outp) {
  const int qb = blockIdx.x * 128;
  const int h  = blockIdx.y;
  const int b  = blockIdx.z;
  __shared__ unsigned short Ks[32][72];      // [key][d], 144B rows (16B aligned)
  __shared__ unsigned short VsT[64][36];     // [d][key] (transposed)
  __shared__ unsigned short Ps[8][16][34];   // per-wave P staging 16x32

  const int tid = threadIdx.x;
  const int lane = tid & 31, wave = tid >> 5;
  const int hi = lane >> 4, nlo = lane & 15;
  const int koff = hi * 8;
  const int qrow_base = qb + wave * 16;
  const float SCALE = 0.03125f;              // C^-0.5 = 1/32 (reference scaling)
  const float NEGBIG = -3.0e38f;

  // Q fragments for this wave's 16 rows (d split into two 32-wide K-chunks)
  FragBF qf[2];
  {
    const size_t rowbase = ((size_t)b * TT + qrow_base + nlo) * (3 * CC) + (size_t)h * DD;
    load_frag16(qf[0], qkv + rowbase, koff);
    load_frag16(qf[1], qkv + rowbase + 32, koff);
  }

  FragF o[4];
  float m[8], l[8];
#pragma unroll
  for (int g = 0; g < 4; ++g)
#pragma unroll
    for (int i = 0; i < 8; ++i) o[g].f[i] = 0.0f;
#pragma unroll
  for (int v = 0; v < 8; ++v) { m[v] = NEGBIG; l[v] = 0.0f; }

  const int jend = qb + 128;
  for (int j0 = 0; j0 < jend; j0 += 32) {
    // K tile 32x64: async b128 copies (one per thread).
    {
      int key = tid >> 3, seg = tid & 7;
      const unsigned short* g =
          qkv + ((size_t)b * TT + j0 + key) * (3 * CC) + CC + (size_t)h * DD + seg * 8;
      async_b128(g, &Ks[key][seg * 8]);
    }
    // V tile 32x64 staged transposed for B-fragment loads.
#pragma unroll
    for (int i = 0; i < 4; ++i) {
      int p = tid + i * 256;
      int key = p >> 5;
      int dcol = (p & 31) * 2;
      unsigned int vv = *(const unsigned int*)(
          qkv + ((size_t)b * TT + j0 + key) * (3 * CC) + 2 * CC + (size_t)h * DD + dcol);
      VsT[dcol][key]     = (unsigned short)(vv & 0xffffu);
      VsT[dcol + 1][key] = (unsigned short)(vv >> 16);
    }
    wait_async0();
    __syncthreads();

    if (j0 <= qrow_base + 15) {  // wave-uniform: tile has >=1 unmasked key
      // S = Q * K^T : two 16x16 subtiles (keys j0..+15, j0+16..+31)
      FragF s0, s1;
#pragma unroll
      for (int i = 0; i < 8; ++i) { s0.f[i] = 0.0f; s1.f[i] = 0.0f; }
#pragma unroll
      for (int c = 0; c < 2; ++c) {
        FragBF kb0, kb1;
        load_frag16(kb0, &Ks[nlo][c * 32], koff);
        load_frag16(kb1, &Ks[16 + nlo][c * 32], koff);
        s0.v = __builtin_amdgcn_wmma_f32_16x16x32_bf16(
            false, qf[c].v, false, kb0.v, (short)0, s0.v, false, false);
        s1.v = __builtin_amdgcn_wmma_f32_16x16x32_bf16(
            false, qf[c].v, false, kb1.v, (short)0, s1.v, false, false);
      }
      // Online softmax per row (rows live in the 16-lane half holding them)
#pragma unroll
      for (int v = 0; v < 8; ++v) {
        int row  = qrow_base + v + hi * 8;
        int col0 = j0 + nlo, col1 = col0 + 16;
        bool ok0 = col0 <= row, ok1 = col1 <= row;
        float a0 = ok0 ? s0.f[v] * SCALE : NEGBIG;
        float a1 = ok1 ? s1.f[v] * SCALE : NEGBIG;
        float mx = fmaxf(a0, a1);
#pragma unroll
        for (int off = 8; off; off >>= 1) mx = fmaxf(mx, __shfl_xor(mx, off, 32));
        float mn = fmaxf(m[v], mx);
        float alpha = __expf(m[v] - mn);
        float p0 = ok0 ? __expf(a0 - mn) : 0.0f;
        float p1 = ok1 ? __expf(a1 - mn) : 0.0f;
        float rs = p0 + p1;
#pragma unroll
        for (int off = 8; off; off >>= 1) rs += __shfl_xor(rs, off, 32);
        l[v] = alpha * l[v] + rs;
        m[v] = mn;
#pragma unroll
        for (int g = 0; g < 4; ++g) o[g].f[v] *= alpha;
        int prow = v + hi * 8;
        Ps[wave][prow][nlo]      = f32_to_bf16(p0);
        Ps[wave][prow][nlo + 16] = f32_to_bf16(p1);
      }
      // Cross-lane LDS RAW within the wave -> explicit CDNA5 split-counter wait
      asm volatile("s_wait_dscnt 0" ::: "memory");
      FragBF pf;
      load_frag16(pf, &Ps[wave][nlo][0], koff);
#pragma unroll
      for (int g = 0; g < 4; ++g) {
        FragBF vb;
        load_frag16(vb, &VsT[g * 16 + nlo][0], koff);
        o[g].v = __builtin_amdgcn_wmma_f32_16x16x32_bf16(
            false, pf.v, false, vb.v, (short)0, o[g].v, false, false);
      }
    }
    __syncthreads();
  }

  // Normalize and write concat-heads output [B*T][C] bf16
#pragma unroll
  for (int v = 0; v < 8; ++v) {
    float inv = 1.0f / l[v];
    int row = qrow_base + v + hi * 8;
#pragma unroll
    for (int g = 0; g < 4; ++g) {
      outp[((size_t)b * TT + row) * CC + h * DD + g * 16 + nlo] =
          f32_to_bf16(o[g].f[v] * inv);
    }
  }
}

// ---------------------------------------------------------------------------
// Host-side orchestration
// ---------------------------------------------------------------------------
extern "C" void kernel_launch(void* const* d_in, const int* in_sizes, int n_in,
                              void* d_out, int out_size, void* d_ws, size_t ws_size,
                              hipStream_t stream) {
  (void)in_sizes; (void)n_in; (void)out_size; (void)ws_size;
  const float* x        = (const float*)d_in[0];
  const float* wq       = (const float*)d_in[1];
  const float* wk       = (const float*)d_in[2];
  const float* wv       = (const float*)d_in[3];
  const float* w_proj   = (const float*)d_in[4];
  const float* b_proj   = (const float*)d_in[5];
  const float* w_ff1    = (const float*)d_in[6];
  const float* b_ff1    = (const float*)d_in[7];
  const float* w_ff2    = (const float*)d_in[8];
  const float* b_ff2    = (const float*)d_in[9];
  const float* ln1s     = (const float*)d_in[10];
  const float* ln1b     = (const float*)d_in[11];
  const float* ln2s     = (const float*)d_in[12];
  const float* ln2b     = (const float*)d_in[13];

  char* p = (char*)d_ws;
  auto take = [&](size_t bytes) -> void* {
    void* r = (void*)p;
    p += (bytes + 255) & ~(size_t)255;
    return r;
  };
  unsigned short* wqkv_b  = (unsigned short*)take((size_t)CC * 3 * CC * 2);
  unsigned short* wproj_b = (unsigned short*)take((size_t)CC * CC * 2);
  unsigned short* wff1_b  = (unsigned short*)take((size_t)CC * 4 * CC * 2);
  unsigned short* wff2_b  = (unsigned short*)take((size_t)4 * CC * CC * 2);
  unsigned short* xn1     = (unsigned short*)take((size_t)MM * CC * 2);
  unsigned short* qkv     = (unsigned short*)take((size_t)MM * 3 * CC * 2);
  unsigned short* attn    = (unsigned short*)take((size_t)MM * CC * 2);
  float*          x1      = (float*)take((size_t)MM * CC * 4);
  unsigned short* xn2     = (unsigned short*)take((size_t)MM * CC * 2);
  unsigned short* h1      = (unsigned short*)take((size_t)MM * 4 * CC * 2);

  // 1) weights -> bf16 (QKV fused)
  pack_qkv_kernel<<<(3 * HH * CC * DD + 255) / 256, 256, 0, stream>>>(wq, wk, wv, wqkv_b);
  convert_kernel<<<(CC * CC + 255) / 256, 256, 0, stream>>>(w_proj, wproj_b, CC * CC);
  convert_kernel<<<(CC * 4 * CC + 255) / 256, 256, 0, stream>>>(w_ff1, wff1_b, CC * 4 * CC);
  convert_kernel<<<(4 * CC * CC + 255) / 256, 256, 0, stream>>>(w_ff2, wff2_b, 4 * CC * CC);

  // 2) LN1 -> bf16
  layernorm_kernel<<<MM, 256, 0, stream>>>(x, ln1s, ln1b, xn1);

  // 3) fused QKV GEMM: [8192,1024] x [1024,3072] -> bf16
  gemm_bf16_kernel<0><<<dim3(3 * CC / 128, MM / 64), 256, 0, stream>>>(
      xn1, wqkv_b, nullptr, nullptr, qkv, MM, 3 * CC, CC);

  // 4) causal flash attention -> bf16 [8192,1024]
  attention_kernel<<<dim3(TT / 128, HH, BB), 256, 0, stream>>>(qkv, attn);

  // 5) output projection + residual: x1 = x + attn @ w_proj + b_proj (f32)
  gemm_bf16_kernel<1><<<dim3(CC / 128, MM / 64), 256, 0, stream>>>(
      attn, wproj_b, b_proj, x, x1, MM, CC, CC);

  // 6) LN2 -> bf16
  layernorm_kernel<<<MM, 256, 0, stream>>>(x1, ln2s, ln2b, xn2);

  // 7) FF1 + bias + ReLU -> bf16 [8192,4096]
  gemm_bf16_kernel<2><<<dim3(4 * CC / 128, MM / 64), 256, 0, stream>>>(
      xn2, wff1_b, b_ff1, nullptr, h1, MM, 4 * CC, CC);

  // 8) FF2 + bias + residual -> f32 output
  gemm_bf16_kernel<1><<<dim3(CC / 128, MM / 64), 256, 0, stream>>>(
      h1, wff2_b, b_ff2, x1, (float*)d_out, MM, CC, 4 * CC);
}
